// OnlyDeformableConv_18416819765755
// MI455X (gfx1250) — compile-verified, matrix operand order checked by reading, not run on hardware
//
#include <hip/hip_runtime.h>

// ---------------------------------------------------------------------------
// OnlyDeformableConv pipeline for MI455X (gfx1250, wave32, WMMA bf16)
//
//  conv1(3->32,5x5)+BN+ReLU      : direct VALU (Cin=3, too skinny for WMMA)
//  conv2(32->32,5x5)+BN+ReLU     : WMMA implicit GEMM (dominant cost)
//  mean over 4 views             : elementwise (commuted BEFORE conv3 - legal,
//                                  conv3 is linear -> 4x less conv3 work)
//  conv3(32->32,3x3)             : WMMA
//  offset conv(32->50->pad64,5x5): WMMA (N padded to 64)
//  deformable conv(5x5)          : bilinear gather in VALU -> WMMA GEMM
//  conv4(32->32,3x3)+BN+ReLU     : WMMA
//  conv5(32->3,3x3)+tanh         : direct VALU
//
// Activations: bf16 NHWC (pixel's 32 channels contiguous = 64B).
// BatchNorm folded into weights/bias at pack time. All WMMA accumulate in f32.
// Weight fragments staged in LDS where <=60KB. Everything is L2 resident.
// ---------------------------------------------------------------------------

typedef __bf16 v16bf __attribute__((ext_vector_type(16)));
typedef float  v8f   __attribute__((ext_vector_type(8)));

static constexpr int HH = 160;
static constexpr int WW = 160;

__device__ __forceinline__ unsigned short f2bf(float f) {
  unsigned u = __float_as_uint(f);
  u += 0x7FFFu + ((u >> 16) & 1u);               // round-to-nearest-even
  return (unsigned short)(u >> 16);
}
__device__ __forceinline__ float bf2f(unsigned short s) {
  return __uint_as_float(((unsigned)s) << 16);
}

// K-index map for 16-bit WMMA fragments (A: lane encodes M, B: lane encodes N):
//   element j<8  -> K = j      + 8*(lane>=16)
//   element j>=8 -> K = j + 8  + 8*(lane>=16)
__device__ __forceinline__ int kmap(int lane, int j) {
  return j + ((j >= 8) ? 8 : 0) + ((lane >= 16) ? 8 : 0);
}

// ---------------------------------------------------------------------------
// Weight packing: OIHW f32 -> WMMA-B bf16 blocks [tap][nhalf][lane][16]
// with optional BN scale folded in.  Cin is always 32 for packed convs.
// ---------------------------------------------------------------------------
__device__ __forceinline__ void pack_b(int idx, const float* __restrict__ w,
                                       int KS, int NH, int realN,
                                       const float* g, const float* vv,
                                       unsigned short* __restrict__ dst) {
  const int j    = idx & 15;
  const int lane = (idx >> 4) & 31;
  const int nh   = (idx >> 9) % NH;
  const int tap  = (idx >> 9) / NH;
  const int o    = nh * 16 + (lane & 15);
  const int c    = kmap(lane, j);
  const int ky   = tap / KS, kx = tap % KS;
  float val = 0.f;
  if (o < realN) {
    val = w[((o * 32 + c) * KS + ky) * KS + kx];
    if (g) val *= g[o] * rsqrtf(vv[o] + 1e-5f);
  }
  dst[idx] = f2bf(val);
}

__global__ __launch_bounds__(256) void pack_weights_kernel(
    const float* __restrict__ w1, const float* __restrict__ b1,
    const float* __restrict__ g1, const float* __restrict__ be1,
    const float* __restrict__ m1, const float* __restrict__ v1,
    const float* __restrict__ w2, const float* __restrict__ b2,
    const float* __restrict__ g2, const float* __restrict__ be2,
    const float* __restrict__ m2, const float* __restrict__ v2,
    const float* __restrict__ w3, const float* __restrict__ b3,
    const float* __restrict__ wo, const float* __restrict__ bo,
    const float* __restrict__ wd, const float* __restrict__ bd,
    const float* __restrict__ w4, const float* __restrict__ b4,
    const float* __restrict__ g4, const float* __restrict__ be4,
    const float* __restrict__ m4, const float* __restrict__ v4,
    unsigned short* w2p, unsigned short* w3p, unsigned short* wop,
    unsigned short* wdp, unsigned short* w4p,
    float* w1f, float* fb1, float* fb2, float* fb3,
    float* fbo, float* fbd, float* fb4)
{
  int tid = blockIdx.x * 256 + threadIdx.x;
  if (tid < 25600) { pack_b(tid, w2, 5, 2, 32, g2, v2, w2p); return; } tid -= 25600;
  if (tid < 9216)  { pack_b(tid, w3, 3, 2, 32, nullptr, nullptr, w3p); return; } tid -= 9216;
  if (tid < 51200) { pack_b(tid, wo, 5, 4, 50, nullptr, nullptr, wop); return; } tid -= 51200;
  if (tid < 25600) { pack_b(tid, wd, 5, 2, 32, nullptr, nullptr, wdp); return; } tid -= 25600;
  if (tid < 9216)  { pack_b(tid, w4, 3, 2, 32, g4, v4, w4p); return; } tid -= 9216;
  if (tid < 2400)  { int o = tid / 75;
                     w1f[tid] = w1[tid] * (g1[o] * rsqrtf(v1[o] + 1e-5f)); return; } tid -= 2400;
  if (tid < 32) { float s = g1[tid] * rsqrtf(v1[tid] + 1e-5f);
                  fb1[tid] = (b1[tid] - m1[tid]) * s + be1[tid]; return; } tid -= 32;
  if (tid < 32) { float s = g2[tid] * rsqrtf(v2[tid] + 1e-5f);
                  fb2[tid] = (b2[tid] - m2[tid]) * s + be2[tid]; return; } tid -= 32;
  if (tid < 32) { fb3[tid] = b3[tid]; return; } tid -= 32;
  if (tid < 64) { fbo[tid] = (tid < 50) ? bo[tid] : 0.f; return; } tid -= 64;
  if (tid < 32) { fbd[tid] = bd[tid]; return; } tid -= 32;
  if (tid < 32) { float s = g4[tid] * rsqrtf(v4[tid] + 1e-5f);
                  fb4[tid] = (b4[tid] - m4[tid]) * s + be4[tid]; return; }
}

// ---------------------------------------------------------------------------
// conv1: 3->32, 5x5, pad 2, BN-folded + ReLU.  f32 NCHW in -> bf16 NHWC out.
// One thread per (img, y, x, oc); oc fastest -> coalesced NHWC stores.
// ---------------------------------------------------------------------------
__global__ __launch_bounds__(256) void conv1_kernel(
    const float* __restrict__ x, const float* __restrict__ w1f,
    const float* __restrict__ fb1, unsigned short* __restrict__ out)
{
  const int tid = blockIdx.x * 256 + threadIdx.x;      // 16*160*160*32 exactly
  const int oc  = tid & 31;
  const int p   = tid >> 5;
  const int xq  = p % WW;
  const int yq  = (p / WW) % HH;
  const int img = p / (WW * HH);
  float sum = fb1[oc];
  const float* wv = w1f + oc * 75;
  #pragma unroll
  for (int c = 0; c < 3; ++c) {
    const float* xc = x + (size_t)(img * 3 + c) * HH * WW;
    #pragma unroll
    for (int ky = 0; ky < 5; ++ky) {
      const int yi = yq + ky - 2;
      if (yi < 0 || yi >= HH) continue;
      #pragma unroll
      for (int kx = 0; kx < 5; ++kx) {
        const int xi = xq + kx - 2;
        if (xi < 0 || xi >= WW) continue;
        sum += xc[yi * WW + xi] * wv[c * 25 + ky * 5 + kx];
      }
    }
  }
  out[tid] = f2bf(fmaxf(sum, 0.f));
}

// ---------------------------------------------------------------------------
// mean over the 4 views (moved before conv3; conv3 is linear)
// ---------------------------------------------------------------------------
__global__ __launch_bounds__(256) void mean4_kernel(
    const unsigned short* __restrict__ in, unsigned short* __restrict__ out)
{
  const int tid  = blockIdx.x * 256 + threadIdx.x;     // 4*25600*32 exactly
  const int cpix = tid % (25600 * 32);
  const int b    = tid / (25600 * 32);
  const size_t st = (size_t)25600 * 32;
  float s = bf2f(in[(size_t)(b * 4 + 0) * st + cpix])
          + bf2f(in[(size_t)(b * 4 + 1) * st + cpix])
          + bf2f(in[(size_t)(b * 4 + 2) * st + cpix])
          + bf2f(in[(size_t)(b * 4 + 3) * st + cpix]);
  out[tid] = f2bf(0.25f * s);
}

// ---------------------------------------------------------------------------
// Generic WMMA implicit-GEMM conv, Cin=32, KSxKS, pad=KS/2.
// One wave computes a 16-pixel x (NH*16)-channel tile. K = taps*32, one
// v_wmma_f32_16x16x32_bf16 per (tap, nhalf).  Weight blocks staged in LDS
// when they fit (<=60KB).  W=160 -> 10 full 16-px tiles per row, no partials.
// ---------------------------------------------------------------------------
template <int KS, int NH, bool RELU, bool OUTF32>
__global__ __launch_bounds__(128) void wmma_conv_kernel(
    const unsigned short* __restrict__ in,   // [imgs][HH][WW][32] bf16
    const unsigned short* __restrict__ wpk,  // [KS*KS][NH][32][16] bf16
    const float* __restrict__ bias,          // [NH*16]
    void* __restrict__ outp, int imgs)
{
  constexpr int PAD = KS / 2;
  constexpr int OC = NH * 16;
  constexpr int WELEMS = KS * KS * NH * 512;
  constexpr bool USE_LDS = (WELEMS * 2 <= 61440);
  __shared__ __align__(32) unsigned short swt[USE_LDS ? WELEMS : 1];
  if constexpr (USE_LDS) {
    for (int i = threadIdx.y * 32 + threadIdx.x; i < WELEMS; i += 128)
      swt[i] = wpk[i];
    __syncthreads();
  }
  const unsigned short* wsrc = USE_LDS ? (const unsigned short*)swt : wpk;

  const int tile = blockIdx.x * 4 + threadIdx.y;
  const int lane = threadIdx.x;
  const int tx   = tile % (WW / 16);
  const int y    = (tile / (WW / 16)) % HH;
  const int img  = tile / ((WW / 16) * HH);
  if (img >= imgs) return;                 // uniform per wave
  const int m  = lane & 15;                // pixel within tile = WMMA M row
  const int hi = lane >> 4;                // fragment half
  const int xq = tx * 16 + m;

  v8f zero = {};
  v8f acc[NH];
  #pragma unroll
  for (int i = 0; i < NH; ++i) acc[i] = zero;

  union AF { v16bf f; uint4 u[2]; };

  #pragma unroll
  for (int ky = 0; ky < KS; ++ky) {
    const int yi = y + ky - PAD;
    const bool yok = (yi >= 0) && (yi < HH);
    #pragma unroll
    for (int kx = 0; kx < KS; ++kx) {
      const int xi = xq + kx - PAD;
      AF a;
      if (yok && (xi >= 0) && (xi < WW)) {
        // NHWC pixel base; channels {8*hi..} and {16+8*hi..} per A layout
        const uint4* base =
            (const uint4*)(in + ((size_t)((size_t)img * HH + yi) * WW + xi) * 32);
        a.u[0] = base[hi];
        a.u[1] = base[2 + hi];
      } else {
        a.u[0] = make_uint4(0, 0, 0, 0);
        a.u[1] = make_uint4(0, 0, 0, 0);
      }
      const int tap = ky * KS + kx;
      #pragma unroll
      for (int nh = 0; nh < NH; ++nh) {
        const v16bf* bw =
            (const v16bf*)(wsrc + (((size_t)tap * NH + nh) * 32 + lane) * 16);
        acc[nh] = __builtin_amdgcn_wmma_f32_16x16x32_bf16(
            false, a.f, false, *bw, (short)0, acc[nh], false, false);
      }
    }
  }

  // C/D layout: N = lane&15, M = vgpr + 8*(lane>=16)
  const int n = lane & 15;
  #pragma unroll
  for (int nh = 0; nh < NH; ++nh) {
    const float bv = bias[nh * 16 + n];
    #pragma unroll
    for (int v = 0; v < 8; ++v) {
      const int Mrow = v + 8 * hi;
      float val = acc[nh][v] + bv;
      if constexpr (RELU) val = fmaxf(val, 0.f);
      const size_t oidx =
          ((size_t)((size_t)img * HH + y) * WW + tx * 16 + Mrow) * OC + nh * 16 + n;
      if constexpr (OUTF32) ((float*)outp)[oidx] = val;
      else                  ((unsigned short*)outp)[oidx] = f2bf(val);
    }
  }
}

// ---------------------------------------------------------------------------
// Deformable conv: per tap, bilinear-gather 16 channels per lane (VALU),
// repack to a bf16 A fragment, then the same WMMA GEMM as the dense convs.
// ---------------------------------------------------------------------------
__device__ __forceinline__ void accum8(float* dst, uint4 u, float wgt) {
  const unsigned a[4] = {u.x, u.y, u.z, u.w};
  #pragma unroll
  for (int k = 0; k < 4; ++k) {
    dst[2 * k]     += wgt * __uint_as_float((a[k] & 0xFFFFu) << 16);
    dst[2 * k + 1] += wgt * __uint_as_float(a[k] & 0xFFFF0000u);
  }
}

__global__ __launch_bounds__(128) void deform_wmma_kernel(
    const unsigned short* __restrict__ xin,  // [4][HH][WW][32] bf16
    const float* __restrict__ offs,          // [4][HH][WW][64] f32 (50 used)
    const unsigned short* __restrict__ wdp,  // [25][2][32][16] bf16
    const float* __restrict__ fbd,           // [32]
    unsigned short* __restrict__ outp)       // [4][HH][WW][32] bf16
{
  constexpr int WELEMS = 25 * 2 * 512;
  __shared__ __align__(32) unsigned short swt[WELEMS];
  for (int i = threadIdx.y * 32 + threadIdx.x; i < WELEMS; i += 128)
    swt[i] = wdp[i];
  __syncthreads();

  const int tile = blockIdx.x * 4 + threadIdx.y;   // 6400 tiles exactly
  const int lane = threadIdx.x;
  const int tx   = tile % 10;
  const int y    = (tile / 10) % HH;
  const int img  = tile / (10 * HH);
  const int m = lane & 15, hi = lane >> 4;
  const int xq = tx * 16 + m;
  const float* offp = offs + ((size_t)((size_t)img * HH + y) * WW + xq) * 64;
  const unsigned short* xb = xin + (size_t)img * HH * WW * 32;

  v8f zero = {};
  v8f acc0 = zero, acc1 = zero;

  for (int t = 0; t < 25; ++t) {
    const int ky = t / 5, kx = t % 5;
    const float dy = offp[2 * t + 0];
    const float dx = offp[2 * t + 1];
    const float py = (float)(y + ky - 2) + dy;
    const float px = (float)(xq + kx - 2) + dx;
    const float yf = floorf(py), xf = floorf(px);
    const float wy = py - yf, wx = px - xf;
    const int y0 = (int)yf, x0 = (int)xf;
    float vals[16];
    #pragma unroll
    for (int i = 0; i < 16; ++i) vals[i] = 0.f;
    #pragma unroll
    for (int cy = 0; cy < 2; ++cy) {
      #pragma unroll
      for (int cx = 0; cx < 2; ++cx) {
        const int yy = y0 + cy, xx = x0 + cx;
        const bool ok = (yy >= 0) && (yy < HH) && (xx >= 0) && (xx < WW);
        float wgt = (cy ? wy : 1.f - wy) * (cx ? wx : 1.f - wx);
        wgt = ok ? wgt : 0.f;                       // reference zeroes OOB corners
        const int yc = min(max(yy, 0), HH - 1);
        const int xc = min(max(xx, 0), WW - 1);
        const uint4* base = (const uint4*)(xb + ((size_t)yc * WW + xc) * 32);
        accum8(vals + 0, base[hi],     wgt);
        accum8(vals + 8, base[2 + hi], wgt);
      }
    }
    union { v16bf f; unsigned int w[8]; } a;
    #pragma unroll
    for (int i = 0; i < 8; ++i)
      a.w[i] = (unsigned)f2bf(vals[2 * i]) | ((unsigned)f2bf(vals[2 * i + 1]) << 16);
    const v16bf* b0 = (const v16bf*)(swt + ((size_t)(t * 2 + 0) * 32 + lane) * 16);
    const v16bf* b1 = (const v16bf*)(swt + ((size_t)(t * 2 + 1) * 32 + lane) * 16);
    acc0 = __builtin_amdgcn_wmma_f32_16x16x32_bf16(false, a.f, false, *b0, (short)0, acc0, false, false);
    acc1 = __builtin_amdgcn_wmma_f32_16x16x32_bf16(false, a.f, false, *b1, (short)0, acc1, false, false);
  }
  const int n = lane & 15;
  const float bv0 = fbd[n], bv1 = fbd[16 + n];
  #pragma unroll
  for (int v = 0; v < 8; ++v) {
    const int Mrow = v + 8 * hi;
    const size_t o = ((size_t)((size_t)img * HH + y) * WW + tx * 16 + Mrow) * 32;
    outp[o + n]      = f2bf(acc0[v] + bv0);
    outp[o + 16 + n] = f2bf(acc1[v] + bv1);
  }
}

// ---------------------------------------------------------------------------
// conv5: 32->3, 3x3, pad 1, + tanh.  bf16 NHWC in -> f32 NCHW out (d_out).
// ---------------------------------------------------------------------------
__global__ __launch_bounds__(256) void conv5_tanh_kernel(
    const unsigned short* __restrict__ xin, const float* __restrict__ w5,
    const float* __restrict__ b5, float* __restrict__ out)
{
  const int tid = blockIdx.x * 256 + threadIdx.x;   // 4*3*160*160 exactly
  const int xq  = tid % WW;
  const int yq  = (tid / WW) % HH;
  const int oc  = (tid / (WW * HH)) % 3;
  const int img = tid / (WW * HH * 3);
  float sum = b5[oc];
  #pragma unroll
  for (int ky = 0; ky < 3; ++ky) {
    const int yi = yq + ky - 1;
    if (yi < 0 || yi >= HH) continue;
    #pragma unroll
    for (int kx = 0; kx < 3; ++kx) {
      const int xi = xq + kx - 1;
      if (xi < 0 || xi >= WW) continue;
      const unsigned short* p = xin + ((size_t)(img * HH + yi) * WW + xi) * 32;
      const float* wv = w5 + oc * 288 + ky * 3 + kx;   // w5[((oc*32+c)*3+ky)*3+kx]
      #pragma unroll
      for (int c = 0; c < 32; ++c) sum += bf2f(p[c]) * wv[c * 9];
    }
  }
  out[tid] = tanhf(sum);   // tid order == [img][oc][y][x] == output layout
}

// ---------------------------------------------------------------------------
extern "C" void kernel_launch(void* const* d_in, const int* in_sizes, int n_in,
                              void* d_out, int out_size, void* d_ws, size_t ws_size,
                              hipStream_t stream)
{
  const float* recon = (const float*)d_in[0];
  const float* w1 = (const float*)d_in[1];  const float* b1 = (const float*)d_in[2];
  const float* g1 = (const float*)d_in[3];  const float* be1= (const float*)d_in[4];
  const float* m1 = (const float*)d_in[5];  const float* v1 = (const float*)d_in[6];
  const float* w2 = (const float*)d_in[7];  const float* b2 = (const float*)d_in[8];
  const float* g2 = (const float*)d_in[9];  const float* be2= (const float*)d_in[10];
  const float* m2 = (const float*)d_in[11]; const float* v2 = (const float*)d_in[12];
  const float* w3 = (const float*)d_in[13]; const float* b3 = (const float*)d_in[14];
  const float* wo = (const float*)d_in[15]; const float* bo = (const float*)d_in[16];
  const float* wd = (const float*)d_in[17]; const float* bd = (const float*)d_in[18];
  const float* w4 = (const float*)d_in[19]; const float* b4 = (const float*)d_in[20];
  const float* g4 = (const float*)d_in[21]; const float* be4= (const float*)d_in[22];
  const float* m4 = (const float*)d_in[23]; const float* v4 = (const float*)d_in[24];
  const float* w5 = (const float*)d_in[25]; const float* b5 = (const float*)d_in[26];

  // ---- workspace layout (~52.7 MB total) ----
  char* ws = (char*)d_ws;
  unsigned short* act1 = (unsigned short*)(ws + 0);          // 26,214,400 B
  unsigned short* act2 = (unsigned short*)(ws + 26214400);   // 26,214,400 B
  char* wp = ws + 52428800;                                  // packed weights
  unsigned short* w2p = (unsigned short*)(wp + 0);           // 51,200 B
  unsigned short* w3p = (unsigned short*)(wp + 51200);       // 18,432 B
  unsigned short* wop = (unsigned short*)(wp + 69632);       // 102,400 B
  unsigned short* wdp = (unsigned short*)(wp + 172032);      // 51,200 B
  unsigned short* w4p = (unsigned short*)(wp + 223232);      // 18,432 B
  float* w1f = (float*)(wp + 241664);                        // 9,600 B
  float* fb1 = (float*)(wp + 251264);
  float* fb2 = (float*)(wp + 251392);
  float* fb3 = (float*)(wp + 251520);
  float* fbo = (float*)(wp + 251648);
  float* fbd = (float*)(wp + 251904);
  float* fb4 = (float*)(wp + 252032);

  // act1 region reused after mean: four [4][160][160][32] bf16 buffers
  unsigned short* xm = act1;                  // mean of views
  unsigned short* x3 = act1 + 3276800;        // conv3 out
  unsigned short* xd = act1 + 2 * 3276800;    // deform out
  unsigned short* x4 = act1 + 3 * 3276800;    // conv4 out
  float* offs = (float*)act2;                 // [4][160][160][64] f32, reuses act2

  const dim3 wb(32, 4);   // 4 wave32 tiles per block

  pack_weights_kernel<<<483, 256, 0, stream>>>(
      w1, b1, g1, be1, m1, v1, w2, b2, g2, be2, m2, v2, w3, b3, wo, bo,
      wd, bd, w4, b4, g4, be4, m4, v4,
      w2p, w3p, wop, wdp, w4p, w1f, fb1, fb2, fb3, fbo, fbd, fb4);

  conv1_kernel<<<51200, 256, 0, stream>>>(recon, w1f, fb1, act1);

  // conv2: 16 imgs * 160 rows * 10 tiles = 25600 tiles
  wmma_conv_kernel<5, 2, true, false><<<6400, wb, 0, stream>>>(act1, w2p, fb2, act2, 16);

  mean4_kernel<<<12800, 256, 0, stream>>>(act2, xm);

  // conv3 on the mean (linearity): 4*160*10 = 6400 tiles
  wmma_conv_kernel<3, 2, false, false><<<1600, wb, 0, stream>>>(xm, w3p, fb3, x3, 4);

  // offset conv: N padded 50->64, f32 NHWC output
  wmma_conv_kernel<5, 4, false, true><<<1600, wb, 0, stream>>>(x3, wop, fbo, offs, 4);

  deform_wmma_kernel<<<1600, wb, 0, stream>>>(x3, offs, wdp, fbd, xd);

  wmma_conv_kernel<3, 2, true, false><<<1600, wb, 0, stream>>>(xd, w4p, fb4, x4, 4);

  conv5_tanh_kernel<<<1200, 256, 0, stream>>>(x4, w5, b5, (float*)d_out);
}
